// FasterRCNN_87617332838800
// MI455X (gfx1250) — matrix operand order repaired and checked by reading, not compiled, not run on hardware
//
#include <hip/hip_runtime.h>
#include <hip/hip_bf16.h>
#include <math.h>

// ---------------------------------------------------------------------------
// CDNA5 (gfx1250) Faster-RCNN forward.  All dense math (13 VGG convs, RPN
// conv, 1x1 heads, fc1/fc2, det heads) runs through v_wmma_f32_16x16x32_f16.
// Conv weights are pre-packed per layer into [Cout][tap][Cin] fp16 so the
// A-tile staging in the hot loop is a single aligned b64 load per thread.
// ---------------------------------------------------------------------------

typedef __attribute__((ext_vector_type(16))) _Float16 v16h;
typedef __attribute__((ext_vector_type(8)))  float    v8f;
typedef __attribute__((ext_vector_type(4)))  float    f4;
typedef __attribute__((ext_vector_type(4)))  _Float16 h4;

__device__ __forceinline__ v8f wmma_f32_16x16x32(v16h a, v16h b, v8f c) {
  // 8 args: (neg_a, A, neg_b, B, c_mod, C, reuse_a, reuse_b)
  return __builtin_amdgcn_wmma_f32_16x16x32_f16(false, a, false, b, (short)0, c,
                                                false, false);
}

__device__ __forceinline__ void transform_box(const float* bb, const float* t,
                                              float* o) {
  float w  = bb[2] - bb[0];
  float h  = bb[3] - bb[1];
  float cx = 0.5f * (bb[0] + bb[2]);
  float cy = 0.5f * (bb[1] + bb[3]);
  float nw = w * expf(t[2]);
  float nh = h * expf(t[3]);
  float ncx = w * t[0] + cx;
  float ncy = h * t[1] + cy;
  o[0] = ncx - 0.5f * nw;
  o[1] = ncy - 0.5f * nh;
  o[2] = ncx + 0.5f * nw;
  o[3] = ncy + 0.5f * nh;
}

// Load the A-fragment (contiguous k = 16*half + e order, same as the B loads) from an LDS tile.
__device__ __forceinline__ v16h load_a_frag(const _Float16* tile, int lane) {
  const int half = lane >> 4;
  const _Float16* As = tile + (lane & 15) * 32;
  v16h a;
#pragma unroll
  for (int e = 0; e < 8; ++e) {
    a[e]     = As[half * 16 + e];       // contiguous 16B -> ds_load_b128
    a[8 + e] = As[half * 16 + 8 + e];
  }
  return a;
}

// ---------------------------------------------------------------------------
// Pack conv weights [Cout][Cin][ks][ks] (f32) -> [Cout][rs][Cin] (f16).
// Tiny one-shot kernel per layer; the packed tile is then re-read by every
// pixel block with coalesced aligned loads.
// ---------------------------------------------------------------------------
__global__ void pack_wt_f16_kernel(const float* __restrict__ w,
                                   _Float16* __restrict__ o,
                                   int Cout, int Cin, int ks2) {
  int idx = blockIdx.x * blockDim.x + threadIdx.x;
  int total = Cout * Cin * ks2;
  if (idx >= total) return;
  int cin = idx % Cin;
  int rs  = (idx / Cin) % ks2;
  int m   = idx / (Cin * ks2);
  o[idx] = (_Float16)w[((size_t)m * Cin + cin) * ks2 + rs];
}

// ---------------------------------------------------------------------------
// Implicit-GEMM convolution as ks*ks shifted GEMMs.  Block = 128 thr =
// 4 waves; block tile = 16 (Cout) x 128 (pixels); each wave owns two 16x16
// WMMA tiles sharing one A fragment.  All global addressing in the hot loop
// is 32-bit offsets from uniform bases.  wt is pre-packed f16 [Cout][rs][Cin].
//   out[Cout, H*W] = act( W x im2col(in) + bias[Cout] )
// ---------------------------------------------------------------------------
__global__ __launch_bounds__(128) void conv_gemm_wmma(
    const float* __restrict__ in, const _Float16* __restrict__ wt,
    const float* __restrict__ bias, float* __restrict__ out,
    int Cin, int Cout, int H, int W, int ks, int relu) {
  const int ks2 = ks * ks;
  const int N   = H * W;
  const int pad = ks >> 1;
  const unsigned uN = (unsigned)N;

  const int lane = threadIdx.x & 31;
  const int wv   = threadIdx.x >> 5;
  const int col  = lane & 15;
  const int half = lane >> 4;

  const int mbase = blockIdx.y * 16;
  const int nb    = blockIdx.x * 128 + wv * 32;
  const int n0    = nb + col;
  const int n1    = n0 + 16;
  const int nc0   = min(n0, N - 1);
  const int nc1   = min(n1, N - 1);
  const int py0 = nc0 / W, px0 = nc0 - py0 * W;
  const int py1 = nc1 / W, px1 = nc1 - py1 * W;

  __shared__ h4 ldsA4[2][128];

  // Weight staging: thread t loads 4 channels of row (mbase + t/8), tap rs.
  const int srow  = threadIdx.x >> 3;  // 0..15
  const int sseg  = threadIdx.x & 7;   // 0..7
  const int warow = min(mbase + srow, Cout - 1);

  v8f acc0 = {}, acc1 = {};
  int pb = 0;
  const int cfull = Cin & ~31;

  for (int r = 0; r < ks; ++r) {
    for (int s = 0; s < ks; ++s) {
      const int rs  = r * ks + s;
      const int iy0 = py0 + r - pad, ix0 = px0 + s - pad;
      const int iy1 = py1 + r - pad, ix1 = px1 + s - pad;
      const bool ok0 = (n0 < N) && iy0 >= 0 && iy0 < H && ix0 >= 0 && ix0 < W;
      const bool ok1 = (n1 < N) && iy1 >= 0 && iy1 < H && ix1 >= 0 && ix1 < W;
      const float m0 = ok0 ? 1.0f : 0.0f;
      const float m1 = ok1 ? 1.0f : 0.0f;
      const unsigned poff0 = ok0 ? (unsigned)(iy0 * W + ix0) : 0u;
      const unsigned poff1 = ok1 ? (unsigned)(iy1 * W + ix1) : 0u;
      // packed row (warow, rs, cin=0..Cin-1), contiguous in cin
      const _Float16* wrow = wt + ((size_t)warow * ks2 + rs) * Cin;

      // ---- branch-free main loop over channel chunks ----
#pragma unroll 2
      for (int c0 = 0; c0 < cfull; c0 += 32) {
        // aligned 8B load (Cin % 32 == 0 in this loop)
        h4 hv = *(const h4*)(wrow + c0 + sseg * 4);
        ldsA4[pb][threadIdx.x] = hv;  // ds_store_b64
        __syncthreads();

        v16h a = load_a_frag((const _Float16*)ldsA4[pb], lane);

        v16h b0, b1;
        const unsigned cbase = (unsigned)(c0 + half * 16) * uN;
#pragma unroll
        for (int e = 0; e < 16; ++e) {  // stride-N, coalesced across lanes
          unsigned co = cbase + (unsigned)e * uN;
          b0[e] = (_Float16)(m0 * in[poff0 + co]);
          b1[e] = (_Float16)(m1 * in[poff1 + co]);
        }
        acc0 = wmma_f32_16x16x32(a, b0, acc0);
        acc1 = wmma_f32_16x16x32(a, b1, acc1);
        pb ^= 1;
      }

      // ---- one-shot guarded tail (only for Cin % 32 != 0, i.e. conv0) ----
      if (cfull < Cin) {
        h4 hv;
#pragma unroll
        for (int j = 0; j < 4; ++j) {
          int cc = cfull + sseg * 4 + j;
          hv[j] = (cc < Cin) ? wrow[cc] : (_Float16)0.0f;
        }
        ldsA4[pb][threadIdx.x] = hv;
        __syncthreads();

        v16h a = load_a_frag((const _Float16*)ldsA4[pb], lane);

        v16h b0, b1;
#pragma unroll
        for (int e = 0; e < 16; ++e) {
          int cc = cfull + half * 16 + e;
          float v0 = 0.0f, v1 = 0.0f;
          if (cc < Cin) {
            unsigned co = (unsigned)cc * uN;
            v0 = m0 * in[poff0 + co];
            v1 = m1 * in[poff1 + co];
          }
          b0[e] = (_Float16)v0;
          b1[e] = (_Float16)v1;
        }
        acc0 = wmma_f32_16x16x32(a, b0, acc0);
        acc1 = wmma_f32_16x16x32(a, b1, acc1);
        pb ^= 1;
      }
    }
  }

#pragma unroll
  for (int r8 = 0; r8 < 8; ++r8) {  // C layout: VGPR r -> M = r + 8*half
    int m = mbase + r8 + half * 8;
    if (m < Cout) {
      float bv = bias[m];
      if (n0 < N) {
        float v = acc0[r8] + bv;
        if (relu) v = fmaxf(v, 0.0f);
        out[(size_t)m * N + n0] = v;
      }
      if (n1 < N) {
        float v = acc1[r8] + bv;
        if (relu) v = fmaxf(v, 0.0f);
        out[(size_t)m * N + n1] = v;
      }
    }
  }
}

// ---------------------------------------------------------------------------
// GEMM: C[M,N] = act( A[M,K] x Bw[N,K]^T + bias[N] )   (fc / det heads)
// Block tile 16x128; per wave two 16x16 tiles.  K must be a multiple of 4
// for the vector path (all call sites have K % 32 == 0).
// ---------------------------------------------------------------------------
__global__ __launch_bounds__(128) void gemm_bias_act_wmma(
    const float* __restrict__ A, const float* __restrict__ Bw,
    const float* __restrict__ bias, float* __restrict__ C,
    int M, int N, int K, int relu) {
  const int lane = threadIdx.x & 31;
  const int wv   = threadIdx.x >> 5;
  const int col  = lane & 15;
  const int half = lane >> 4;

  const int mbase = blockIdx.y * 16;
  const int nb    = blockIdx.x * 128 + wv * 32;
  const int n0    = nb + col;
  const int n1    = n0 + 16;
  const float* bp0 = Bw + (size_t)min(n0, N - 1) * K + half * 16;
  const float* bp1 = Bw + (size_t)min(n1, N - 1) * K + half * 16;

  __shared__ h4 ldsA4[2][128];

  const int srow = threadIdx.x >> 3;
  const int sseg = threadIdx.x & 7;
  const float* ap = A + (size_t)min(mbase + srow, M - 1) * K + sseg * 4;

  v8f acc0 = {}, acc1 = {};
  int pb = 0;
  const int kfull = K & ~31;

#pragma unroll 2
  for (int kb = 0; kb < kfull; kb += 32) {
    f4 av = *(const f4*)(ap + kb);  // global_load_b128
    h4 hv;
#pragma unroll
    for (int j = 0; j < 4; ++j) hv[j] = (_Float16)av[j];
    ldsA4[pb][threadIdx.x] = hv;
    __syncthreads();

    v16h a = load_a_frag((const _Float16*)ldsA4[pb], lane);

    __builtin_prefetch(bp0 + kb + 256, 0, 1);  // global_prefetch_b8

    v16h b0, b1;
    const f4* q0 = (const f4*)(bp0 + kb);
    const f4* q1 = (const f4*)(bp1 + kb);
#pragma unroll
    for (int v4i = 0; v4i < 4; ++v4i) {
      f4 t0 = q0[v4i];
      f4 t1 = q1[v4i];
#pragma unroll
      for (int j = 0; j < 4; ++j) {
        b0[v4i * 4 + j] = (_Float16)t0[j];
        b1[v4i * 4 + j] = (_Float16)t1[j];
      }
    }
    acc0 = wmma_f32_16x16x32(a, b0, acc0);
    acc1 = wmma_f32_16x16x32(a, b1, acc1);
    pb ^= 1;
  }

  if (kfull < K) {  // uniform guarded tail (not taken by current call sites)
    h4 hv;
#pragma unroll
    for (int j = 0; j < 4; ++j) {
      int kk = kfull + sseg * 4 + j;
      hv[j] = (_Float16)((kk < K) ? ap[kk - sseg * 4] : 0.0f);
    }
    ldsA4[pb][threadIdx.x] = hv;
    __syncthreads();
    v16h a = load_a_frag((const _Float16*)ldsA4[pb], lane);
    v16h b0, b1;
#pragma unroll
    for (int e = 0; e < 16; ++e) {
      int k = kfull + half * 16 + e;
      // bp0/bp1 already include the +half*16 bias, so subtract it back out.
      b0[e] = (_Float16)((k < K) ? bp0[k - half * 16] : 0.0f);
      b1[e] = (_Float16)((k < K) ? bp1[k - half * 16] : 0.0f);
    }
    acc0 = wmma_f32_16x16x32(a, b0, acc0);
    acc1 = wmma_f32_16x16x32(a, b1, acc1);
  }

#pragma unroll
  for (int r8 = 0; r8 < 8; ++r8) {
    int m = mbase + r8 + half * 8;
    if (m < M) {
      if (n0 < N) {
        float v = acc0[r8] + bias[n0];
        if (relu) v = fmaxf(v, 0.0f);
        C[(size_t)m * N + n0] = v;
      }
      if (n1 < N) {
        float v = acc1[r8] + bias[n1];
        if (relu) v = fmaxf(v, 0.0f);
        C[(size_t)m * N + n1] = v;
      }
    }
  }
}

// ---------------------------------------------------------------------------
// 2x2 max pool, stride 2 (VALID).
// ---------------------------------------------------------------------------
__global__ void maxpool2x2_kernel(const float* __restrict__ in,
                                  float* __restrict__ out, int C, int H, int W) {
  int Ho = H >> 1, Wo = W >> 1;
  int total = C * Ho * Wo;
  int idx = blockIdx.x * blockDim.x + threadIdx.x;
  if (idx >= total) return;
  int c = idx / (Ho * Wo);
  int rem = idx - c * (Ho * Wo);
  int y = rem / Wo;
  int x = rem - y * Wo;
  const float* p = in + ((size_t)c * H + 2 * y) * W + 2 * x;
  out[idx] = fmaxf(fmaxf(p[0], p[1]), fmaxf(p[W], p[W + 1]));
}

// ---------------------------------------------------------------------------
// RPN post: per-anchor softmax, box readout, anchors, proposal transform,
// inside test, score.
// ---------------------------------------------------------------------------
__global__ void rpn_post_kernel(const float* __restrict__ clsRaw,
                                const float* __restrict__ boxRaw,
                                float* __restrict__ outCls,
                                float* __restrict__ outSm,
                                float* __restrict__ outBox,
                                float* __restrict__ props,
                                float* __restrict__ score,
                                int Hf, int Wf, float imW, float imH) {
  int A = Hf * Wf * 9;
  int a = blockIdx.x * blockDim.x + threadIdx.x;
  if (a >= A) return;
  int pix = a / 9;
  int k   = a - pix * 9;
  int y   = pix / Wf;
  int x   = pix - y * Wf;
  int NP  = Hf * Wf;

  float c0 = clsRaw[(2 * k) * NP + pix];
  float c1 = clsRaw[(2 * k + 1) * NP + pix];
  float mx = fmaxf(c0, c1);
  float l0 = c0 - mx, l1 = c1 - mx;
  outCls[a * 2 + 0] = l0;
  outCls[a * 2 + 1] = l1;
  float e0 = expf(l0), e1 = expf(l1);
  float inv = 1.0f / (e0 + e1);
  outSm[a * 2 + 0] = e0 * inv;
  float p1 = e1 * inv;
  outSm[a * 2 + 1] = p1;

  float t[4];
#pragma unroll
  for (int j = 0; j < 4; ++j) {
    t[j] = boxRaw[(4 * k + j) * NP + pix];
    outBox[a * 4 + j] = t[j];
  }

  int ai = k / 3, ri = k - ai * 3;
  float side  = 128.0f * (float)(1 << ai);
  float area  = side * side;
  float ratio = (ri == 0) ? 0.5f : (ri == 1 ? 1.0f : 2.0f);
  float wA = sqrtf(area / ratio);
  float hA = wA * ratio;
  float cx = (float)x * 16.0f;
  float cy = (float)y * 16.0f;
  float bb[4] = {cx - 0.5f * wA, cy - 0.5f * hA, cx + 0.5f * wA, cy + 0.5f * hA};
  float pr[4];
  transform_box(bb, t, pr);
#pragma unroll
  for (int j = 0; j < 4; ++j) props[a * 4 + j] = pr[j];
  bool inside = (pr[0] >= 0.0f) && (pr[1] >= 0.0f) && (pr[2] <= imW) && (pr[3] <= imH);
  score[a] = inside ? p1 : -1.0f;
}

// ---------------------------------------------------------------------------
// Deterministic top-K via exact rank (ties -> lower index, like lax.top_k).
// ---------------------------------------------------------------------------
__global__ void topk_select_kernel(const float* __restrict__ score,
                                   const float* __restrict__ props,
                                   float* __restrict__ rois, int A, int TK) {
  __shared__ float sc[256];
  int i = blockIdx.x * blockDim.x + threadIdx.x;
  float si = (i < A) ? score[i] : -1e30f;
  int rank = 0;
  for (int base = 0; base < A; base += 256) {
    int cnt = min(256, A - base);
    for (int tloc = threadIdx.x; tloc < cnt; tloc += blockDim.x)
      sc[tloc] = score[base + tloc];
    __syncthreads();
    if (i < A) {
      for (int tloc = 0; tloc < cnt; ++tloc) {
        float sj = sc[tloc];
        int j = base + tloc;
        rank += (sj > si) || (sj == si && j < i);
      }
    }
    __syncthreads();
  }
  if (i < A && rank < TK) {
    rois[rank * 4 + 0] = props[i * 4 + 0];
    rois[rank * 4 + 1] = props[i * 4 + 1];
    rois[rank * 4 + 2] = props[i * 4 + 2];
    rois[rank * 4 + 3] = props[i * 4 + 3];
  }
}

// ---------------------------------------------------------------------------
// ROI max-pool (POOL=7, MAXB=8), matching reference bin/clip semantics.
// Branch-free 8x8 scan: clamped loads + select.  pooled: [roi][c][ph][pw].
// ---------------------------------------------------------------------------
__global__ void roi_pool_kernel(const float* __restrict__ fmap,
                                const float* __restrict__ rois,
                                float* __restrict__ pooled,
                                int C, int Hf, int Wf) {
  int total = 512 * C * 49;
  int idx = blockIdx.x * blockDim.x + threadIdx.x;
  if (idx >= total) return;
  int bin = idx % 49;
  int c   = (idx / 49) % C;
  int roi = idx / (49 * C);
  int ph = bin / 7, pw = bin - ph * 7;

  float x1 = rintf(rois[roi * 4 + 0] * 0.0625f);
  float y1 = rintf(rois[roi * 4 + 1] * 0.0625f);
  float x2 = rintf(rois[roi * 4 + 2] * 0.0625f);
  float y2 = rintf(rois[roi * 4 + 3] * 0.0625f);
  float rw = fmaxf(x2 - x1 + 1.0f, 1.0f);
  float rh = fmaxf(y2 - y1 + 1.0f, 1.0f);
  float bw = rw * (1.0f / 7.0f);
  float bh = rh * (1.0f / 7.0f);

  int hs = (int)fminf(fmaxf(floorf((float)ph * bh) + y1, 0.0f), (float)Hf);
  int he = (int)fminf(fmaxf(ceilf((float)(ph + 1) * bh) + y1, 0.0f), (float)Hf);
  int wsv = (int)fminf(fmaxf(floorf((float)pw * bw) + x1, 0.0f), (float)Wf);
  int we = (int)fminf(fmaxf(ceilf((float)(pw + 1) * bw) + x1, 0.0f), (float)Wf);

  const float* fc = fmap + (size_t)c * Hf * Wf;
  float m = -3.402823466e+38f;
  int any = 0;
#pragma unroll
  for (int dy = 0; dy < 8; ++dy) {
    int yy = hs + dy;
    int ysc = min(yy, Hf - 1);
    bool vy = yy < he;
#pragma unroll
    for (int dx = 0; dx < 8; ++dx) {
      int xx = wsv + dx;
      int xsc = min(xx, Wf - 1);
      bool v = vy && (xx < we);
      float val = fc[ysc * Wf + xsc];
      m = v ? fmaxf(m, val) : m;
      any |= (int)v;
    }
  }
  pooled[idx] = any ? m : 0.0f;
}

// ---------------------------------------------------------------------------
// Detection epilogue: softmax over 21 classes, argmax (first max), select
// per-class regression, transform with ROI, emit locs + pred.
// ---------------------------------------------------------------------------
__global__ void det_post_kernel(const float* __restrict__ logits,
                                const float* __restrict__ breg,
                                const float* __restrict__ rois,
                                float* __restrict__ outSm,
                                float* __restrict__ outLocs,
                                float* __restrict__ outPred) {
  int i = blockIdx.x * blockDim.x + threadIdx.x;
  if (i >= 512) return;
  const float* L = logits + i * 21;
  float mx = L[0];
  int bi = 0;
  for (int j = 1; j < 21; ++j) {
    float v = L[j];
    if (v > mx) { mx = v; bi = j; }
  }
  float e[21];
  float s = 0.0f;
  for (int j = 0; j < 21; ++j) { e[j] = expf(L[j] - mx); s += e[j]; }
  float inv = 1.0f / s;
  for (int j = 0; j < 21; ++j) outSm[i * 21 + j] = e[j] * inv;
  float t[4];
  for (int j = 0; j < 4; ++j) t[j] = breg[i * 84 + bi * 4 + j];
  float o[4];
  transform_box(&rois[i * 4], t, o);
  for (int j = 0; j < 4; ++j) outLocs[i * 4 + j] = o[j];
  outPred[i] = (float)bi;
}

// ---------------------------------------------------------------------------
// Host-side orchestration.
// ---------------------------------------------------------------------------
extern "C" void kernel_launch(void* const* d_in, const int* in_sizes, int n_in,
                              void* d_out, int out_size, void* d_ws,
                              size_t ws_size, hipStream_t stream) {
  (void)in_sizes; (void)n_in; (void)out_size; (void)ws_size;

  const float* image = (const float*)d_in[0];
  const float* convw[13];
  const float* convb[13];
  for (int i = 0; i < 13; ++i) {
    convw[i] = (const float*)d_in[1 + i];
    convb[i] = (const float*)d_in[14 + i];
  }
  const float* fc1_w = (const float*)d_in[27];
  const float* fc1_b = (const float*)d_in[28];
  const float* fc2_w = (const float*)d_in[29];
  const float* fc2_b = (const float*)d_in[30];
  const float* rpn_conv_w = (const float*)d_in[31];
  const float* rpn_conv_b = (const float*)d_in[32];
  const float* rpn_cls_w  = (const float*)d_in[33];
  const float* rpn_cls_b  = (const float*)d_in[34];
  const float* rpn_box_w  = (const float*)d_in[35];
  const float* rpn_box_b  = (const float*)d_in[36];
  const float* det_cls_w  = (const float*)d_in[37];
  const float* det_cls_b  = (const float*)d_in[38];
  const float* det_box_w  = (const float*)d_in[39];
  const float* det_box_b  = (const float*)d_in[40];
  float* out = (float*)d_out;

  // Output layout (floats): rpn_cls | rpn_softmax | rpn_box | cls_logits |
  //                         cls_softmax | locs | pred
  const int O_RPN_CLS = 0;      // 6912*2
  const int O_RPN_SM  = 13824;  // 6912*2
  const int O_RPN_BOX = 27648;  // 6912*4
  const int O_LOGITS  = 55296;  // 512*21
  const int O_CLS_SM  = 66048;  // 512*21
  const int O_LOCS    = 76800;  // 512*4
  const int O_PRED    = 78848;  // 512

  float* ws = (float*)d_ws;
  size_t off = 0;
  auto carve = [&](size_t n) {
    float* p = ws + off;
    off += (n + 255) & ~(size_t)255;
    return p;
  };
  float* bufA      = carve((size_t)64 * 384 * 512);  // conv ping
  float* bufB      = carve((size_t)64 * 384 * 512);  // conv pong
  float* rpnH      = carve((size_t)512 * 24 * 32);   // RPN hidden
  float* rpnClsRaw = carve((size_t)18 * 24 * 32);
  float* rpnBoxRaw = carve((size_t)36 * 24 * 32);
  float* props     = carve((size_t)6912 * 4);
  float* scores    = carve((size_t)6912);
  float* rois      = carve((size_t)512 * 4);
  float* pooled    = carve((size_t)512 * 25088);
  float* x1b       = carve((size_t)512 * 4096);
  float* x2b       = carve((size_t)512 * 4096);
  float* bregb     = carve((size_t)512 * 84);
  // packed f16 conv weights, reused per layer (max 512*512*9 halfs)
  _Float16* wtpack = (_Float16*)carve((size_t)512 * 512 * 9 / 2 + 256);

  auto launch_conv = [&](const float* src, const float* w, const float* b,
                         float* dst, int Cin, int Cout, int H, int W, int ks,
                         int relu) {
    int ks2 = ks * ks;
    int tot = Cout * Cin * ks2;
    pack_wt_f16_kernel<<<dim3((tot + 255) / 256), dim3(256), 0, stream>>>(
        w, wtpack, Cout, Cin, ks2);
    int N = H * W;
    dim3 g((N + 127) / 128, (Cout + 15) / 16);
    conv_gemm_wmma<<<g, dim3(128), 0, stream>>>(src, wtpack, b, dst, Cin, Cout,
                                                H, W, ks, relu);
  };

  // ---- VGG-16 feature extractor ----
  const int cfg[17] = {64, 64, -1, 128, 128, -1, 256, 256, 256, -1,
                       512, 512, 512, -1, 512, 512, 512};
  const float* x = image;
  float* bufs[2] = {bufA, bufB};
  int bi = 0, wi = 0, H = 384, W = 512, Cin = 3;
  for (int li = 0; li < 17; ++li) {
    if (cfg[li] < 0) {
      int Ho = H / 2, Wo = W / 2;
      int total = Cin * Ho * Wo;
      maxpool2x2_kernel<<<dim3((total + 255) / 256), dim3(256), 0, stream>>>(
          x, bufs[bi], Cin, H, W);
      x = bufs[bi]; bi ^= 1; H = Ho; W = Wo;
    } else {
      int Cout = cfg[li];
      launch_conv(x, convw[wi], convb[wi], bufs[bi], Cin, Cout, H, W, 3, 1);
      x = bufs[bi]; bi ^= 1; Cin = Cout; ++wi;
    }
  }
  const float* fmap = x;  // 512 x 24 x 32

  // ---- RPN head ----
  launch_conv(fmap, rpn_conv_w, rpn_conv_b, rpnH, 512, 512, 24, 32, 3, 1);
  launch_conv(rpnH, rpn_cls_w, rpn_cls_b, rpnClsRaw, 512, 18, 24, 32, 1, 0);
  launch_conv(rpnH, rpn_box_w, rpn_box_b, rpnBoxRaw, 512, 36, 24, 32, 1, 0);

  // ---- Anchors / proposals / scores ----
  rpn_post_kernel<<<dim3((6912 + 127) / 128), dim3(128), 0, stream>>>(
      rpnClsRaw, rpnBoxRaw, out + O_RPN_CLS, out + O_RPN_SM, out + O_RPN_BOX,
      props, scores, 24, 32, 512.0f, 384.0f);

  // ---- Top-512 selection ----
  topk_select_kernel<<<dim3((6912 + 127) / 128), dim3(128), 0, stream>>>(
      scores, props, rois, 6912, 512);

  // ---- ROI max-pool ----
  {
    int total = 512 * 512 * 49;
    roi_pool_kernel<<<dim3((total + 255) / 256), dim3(256), 0, stream>>>(
        fmap, rois, pooled, 512, 24, 32);
  }

  // ---- FC + detection heads (WMMA GEMMs) ----
  gemm_bias_act_wmma<<<dim3(32, 32), dim3(128), 0, stream>>>(
      pooled, fc1_w, fc1_b, x1b, 512, 4096, 25088, 1);
  gemm_bias_act_wmma<<<dim3(32, 32), dim3(128), 0, stream>>>(
      x1b, fc2_w, fc2_b, x2b, 512, 4096, 4096, 1);
  gemm_bias_act_wmma<<<dim3(1, 32), dim3(128), 0, stream>>>(
      x2b, det_cls_w, det_cls_b, out + O_LOGITS, 512, 21, 4096, 0);
  gemm_bias_act_wmma<<<dim3(1, 32), dim3(128), 0, stream>>>(
      x2b, det_box_w, det_box_b, bregb, 512, 84, 4096, 0);

  // ---- Detection epilogue ----
  det_post_kernel<<<dim3(4), dim3(128), 0, stream>>>(
      out + O_LOGITS, bregb, rois, out + O_CLS_SM, out + O_LOCS, out + O_PRED);
}
